// Encoder_35725537968324
// MI455X (gfx1250) — matrix operand order repaired and checked by reading, not compiled, not run on hardware
//
#include <hip/hip_runtime.h>
#include <hip/hip_bf16.h>

typedef __attribute__((ext_vector_type(16))) _Float16 v16h;
typedef __attribute__((ext_vector_type(8)))  float    v8f;

// ---------------------------------------------------------------- utilities
__global__ void k_fill(float* __restrict__ p, float v, int n) {
    int i = blockIdx.x * blockDim.x + threadIdx.x;
    if (i < n) p[i] = v;
}

__global__ void k_deg_count(float* __restrict__ deg,
                            const long long* __restrict__ col, int e) {
    int i = blockIdx.x * blockDim.x + threadIdx.x;
    if (i < e) atomicAdd(&deg[col[i]], 1.0f);
}

__global__ void k_rsqrt(float* __restrict__ p, int n) {
    int i = blockIdx.x * blockDim.x + threadIdx.x;
    if (i < n) p[i] = rsqrtf(p[i]);
}

// -------------------------------------------------- operand packing (K = 128)
// B: f32 row-major [128 x Ncol] -> f16 fragments.
// Record ((nt*4 + kc)*32 + lane)*16 + j holds B[kc*32 + (lane>=16?8:0) + (j&7) + ((j>>3)<<4)][nt*16 + (lane&15)].
__global__ void k_pack_b(const float* __restrict__ W,
                         _Float16* __restrict__ Bp, int Ncol) {
    int idx = blockIdx.x * blockDim.x + threadIdx.x;
    if (idx >= Ncol * 128) return;
    int j    = idx & 15;
    int lane = (idx >> 4) & 31;
    int kc   = (idx >> 9) & 3;
    int nt   = idx >> 11;
    int n    = (nt << 4) + (lane & 15);
    int k    = kc * 32 + ((lane >> 4) << 3) + (j & 7) + ((j >> 3) << 4);
    Bp[idx] = (_Float16)W[(size_t)k * Ncol + n];
}

// A: f32 row-major [M x 128] -> f16 fragments, zero-padded past M
// (optionally fused ReLU).  Record ((mt*4 + kc)*32 + lane)*16 + j holds
//   A[mt*16 + (lane&15)][kc*32 + (lane>=16?8:0) + (j&7) + ((j>>3)<<4)].
__global__ void k_pack_a(const float* __restrict__ S,
                         _Float16* __restrict__ Ap, int M, int total, int relu) {
    int idx = blockIdx.x * blockDim.x + threadIdx.x;
    if (idx >= total) return;
    int j    = idx & 15;
    int lane = (idx >> 4) & 31;
    int kc   = (idx >> 9) & 3;
    int mt   = idx >> 11;
    int mrow = (mt << 4) + (lane & 15);
    int k    = kc * 32 + ((lane >> 4) << 3) + (j & 7) + ((j >> 3) << 4);
    float v  = 0.0f;
    if (mrow < M) v = S[(size_t)mrow * 128 + k];
    if (relu)     v = fmaxf(v, 0.0f);
    Ap[idx] = (_Float16)v;
}

// ---------------------------------------------------------------- WMMA GEMM
// C[Mpad x Ncol] = A[Mpad x 128] * B[128 x Ncol], pre-packed f16 fragments,
// Mpad a multiple of 64.  One wave computes a 64x16 block, fully unrolled:
// 16 straight-line WMMAs with EXEC all-ones (no divergence anywhere).
__global__ void k_gemm_wmma(const _Float16* __restrict__ Ap,
                            const _Float16* __restrict__ Bp,
                            float* __restrict__ C, int Ncol) {
    const int lane  = threadIdx.x & 31;
    const int wave  = threadIdx.x >> 5;
    const int ntil  = Ncol >> 4;
    const int gtile = blockIdx.x * (blockDim.x >> 5) + wave;
    const int mb    = gtile / ntil;          // 64-row block index
    const int nt    = gtile - mb * ntil;
    const int m0    = mb << 6;
    const int mt0   = mb << 2;               // first 16-row subtile
    const int l15   = lane & 15;
    const int hi    = lane >> 4;
    const int ncol  = (nt << 4) + l15;

    v8f acc[4] = {v8f{}, v8f{}, v8f{}, v8f{}};

    #pragma unroll
    for (int kc = 0; kc < 4; ++kc) {
        v16h b = *(const v16h*)(Bp + ((((size_t)nt * 4 + kc) * 32 + lane) << 4));
        #pragma unroll
        for (int s = 0; s < 4; ++s) {
            v16h a = *(const v16h*)(Ap +
                        ((((size_t)(mt0 + s) * 4 + kc) * 32 + lane) << 4));
            acc[s] = __builtin_amdgcn_wmma_f32_16x16x32_f16(
                         false, a, false, b, (short)0, acc[s], false, false);
        }
    }

    // f32 C/D layout: VGPR r -> M = r (lanes 0-15) / 8+r (lanes 16-31), N = l15
    const int rb = hi ? 8 : 0;
    #pragma unroll
    for (int s = 0; s < 4; ++s) {
        const int mrow = m0 + s * 16 + rb;
        #pragma unroll
        for (int r = 0; r < 8; ++r)
            C[(size_t)(mrow + r) * Ncol + ncol] = acc[s][r];
    }
}

// ---------------------------------------------------------------- aggregation
// Aout[i,c] = bias[c] + 2 * dis[i]^2 * T[i,c]   (duplicate self-loop pair)
__global__ void k_agg_init(float* __restrict__ Aout,
                           const float* __restrict__ T,
                           const float* __restrict__ dis,
                           const float* __restrict__ bias,
                           int n, int F) {
    int idx = blockIdx.x * blockDim.x + threadIdx.x;
    if (idx >= n * F) return;
    int i = idx / F;
    int c = idx - i * F;
    float d = dis[i];
    Aout[idx] = bias[c] + 2.0f * d * d * T[idx];
}

// One wave per edge: Aout[col] += dis[row]*dis[col] * T[row]  (f32 L2 atomics)
__global__ void k_agg_edges(const long long* __restrict__ row,
                            const long long* __restrict__ col,
                            const float* __restrict__ dis,
                            const float* __restrict__ T,
                            float* __restrict__ Aout,
                            int e, int F) {
    int gt   = blockIdx.x * blockDim.x + threadIdx.x;
    int wid  = gt >> 5;
    int lane = gt & 31;
    if (wid >= e) return;
    long long r = row[wid];
    long long c = col[wid];
    float nrm = dis[r] * dis[c];
    const float* src = T + (size_t)r * F;
    float*       dst = Aout + (size_t)c * F;
    for (int f = lane * 4; f < F; f += 128) {
        const float4 v = *(const float4*)(src + f);
        atomicAdd(dst + f + 0, nrm * v.x);
        atomicAdd(dst + f + 1, nrm * v.y);
        atomicAdd(dst + f + 2, nrm * v.z);
        atomicAdd(dst + f + 3, nrm * v.w);
    }
}

// ---------------------------------------------------------------- launcher
static inline size_t alignup(size_t x) { return (x + 255) & ~(size_t)255; }
static inline int cdiv(long long a, long long b) { return (int)((a + b - 1) / b); }

extern "C" void kernel_launch(void* const* d_in, const int* in_sizes, int n_in,
                              void* d_out, int out_size, void* d_ws, size_t ws_size,
                              hipStream_t stream) {
    (void)n_in; (void)out_size; (void)ws_size;

    const float*     x   = (const float*)d_in[0];
    const long long* ei  = (const long long*)d_in[1];   // int64 [2, E]
    const float*     W1  = (const float*)d_in[2];
    const float*     b1  = (const float*)d_in[3];
    const float*     W2  = (const float*)d_in[4];
    const float*     b2  = (const float*)d_in[5];
    float*           out = (float*)d_out;

    const int E   = in_sizes[1] / 2;
    const int HID = in_sizes[3];                 // 128
    const int OUT = in_sizes[5];                 // 64
    const int IN  = in_sizes[2] / HID;           // 128
    const int N   = in_sizes[0] / IN;            // 100000

    const long long* erow = ei;
    const long long* ecol = ei + E;

    const int M64    = cdiv(N, 64);              // 64-row GEMM blocks
    const int Mpad   = M64 * 64;                 // padded row count
    const int aTotal = M64 * 4 * 2048;           // packed-A halves (K=128)

    // workspace carve (Hp aliases Ap; T2 aliases T1); T/A padded to Mpad rows
    char* p = (char*)d_ws;
    float*    dis = (float*)p;    p += alignup((size_t)N * 4);
    _Float16* Ap  = (_Float16*)p; p += alignup((size_t)aTotal * 2);
    _Float16* Bp1 = (_Float16*)p; p += alignup((size_t)IN * HID * 2);
    _Float16* Bp2 = (_Float16*)p; p += alignup((size_t)HID * OUT * 2);
    float*    T1  = (float*)p;    p += alignup((size_t)Mpad * HID * 4);
    float*    A1  = (float*)p;    p += alignup((size_t)Mpad * HID * 4);
    _Float16* Hp  = Ap;   // x fragments dead after GEMM1
    float*    T2  = T1;   // T1 dead after ReLU repack (Mpad*OUT <= Mpad*HID)

    const int TB = 256;

    // 1) degree (dup self-loops => start at 2), then d^-1/2
    k_fill     <<<cdiv(N, TB), TB, 0, stream>>>(dis, 2.0f, N);
    k_deg_count<<<cdiv(E, TB), TB, 0, stream>>>(dis, ecol, E);
    k_rsqrt    <<<cdiv(N, TB), TB, 0, stream>>>(dis, N);

    // 2) pack operands into WMMA fragment order (fused f32->f16, zero-padded)
    k_pack_a<<<cdiv(aTotal, TB), TB, 0, stream>>>(x,  Ap,  N, aTotal, 0);
    k_pack_b<<<cdiv((long long)IN * HID, TB), TB, 0, stream>>>(W1, Bp1, HID);
    k_pack_b<<<cdiv((long long)HID * OUT, TB), TB, 0, stream>>>(W2, Bp2, OUT);

    // 3) layer 1: T1 = x @ W1
    k_gemm_wmma<<<cdiv((long long)M64 * (HID / 16), 8), TB, 0, stream>>>(Ap, Bp1, T1, HID);
    // 4) aggregate: A1 = b1 + selfloops + edge scatter
    k_agg_init <<<cdiv((long long)N * HID, TB), TB, 0, stream>>>(A1, T1, dis, b1, N, HID);
    k_agg_edges<<<cdiv((long long)E * 32, TB), TB, 0, stream>>>(erow, ecol, dis, T1, A1, E, HID);
    // 5) H = relu(A1) packed as f16 fragments for GEMM2
    k_pack_a<<<cdiv(aTotal, TB), TB, 0, stream>>>(A1, Hp, N, aTotal, 1);

    // 6) layer 2: T2 = H @ W2
    k_gemm_wmma<<<cdiv((long long)M64 * (OUT / 16), 8), TB, 0, stream>>>(Hp, Bp2, T2, OUT);
    // 7) out = b2 + selfloops + edge scatter
    k_agg_init <<<cdiv((long long)N * OUT, TB), TB, 0, stream>>>(out, T2, dis, b2, N, OUT);
    k_agg_edges<<<cdiv((long long)E * 32, TB), TB, 0, stream>>>(erow, ecol, dis, T2, out, E, OUT);
}
